// CenterLossNet_75900662055406
// MI455X (gfx1250) — compile-verified, matrix operand order checked by reading, not run on hardware
//
#include <hip/hip_runtime.h>
#include <math.h>

// ---------------------------------------------------------------------------
// CenterLoss on MI455X (gfx1250, wave32, WMMA + TDM)
//
// Core: 8192 x 10000 x 512 GEMM -> v_wmma_f32_16x16x32_bf16, bf16x3 split
// (hi + lo residual planes) for ~f32 accuracy.
// Data path: B (centers) panel staged in LDS via Tensor Data Mover
// (tensor_load_to_lds, 160KB/workgroup of CDNA5's 320KB LDS), shared by the
// 8 waves of the workgroup -> 8x less global B traffic; WMMA operands fed by
// ds_load_b128 (B) + global_load_b128 (A).
// Deterministic: per-(row, colgroup) partials + fixed-order reduction.
// ---------------------------------------------------------------------------

typedef __bf16 bf16_t;
typedef __attribute__((ext_vector_type(16))) __bf16 v16bf;
typedef __attribute__((ext_vector_type(8)))  __bf16 v8bf;
typedef __attribute__((ext_vector_type(8)))  float  v8f;
typedef __attribute__((ext_vector_type(4)))  unsigned int v4u;
typedef __attribute__((ext_vector_type(8)))  int v8i;
typedef __attribute__((ext_vector_type(4)))  int v4i;

#define N_ROWS 8192
#define N_CLS  10000
#define DIM    512
#define SCALE  5.0f
#define NTILE  5               // column tiles (of 16) per wave -> 80 cols
#define NGROUP 125             // 10000 / 80
#define MTILES (N_ROWS / 16)   // 512
#define WAVES_PER_BLK 8        // 8 M-tiles per workgroup share one B panel
#define BPANEL_ELEMS (NTILE * 16 * DIM)          // 80 x 512 bf16
#define BPANEL_BYTES (BPANEL_ELEMS * 2)          // 81920 B per plane
#define LDS_BYTES    (2 * BPANEL_BYTES)          // hi + lo = 163840 B

// ---------------------------------------------------------------------------
// Phase 1: L2-normalize rows, emit bf16 hi/lo planes + ||normalized||^2.
// ---------------------------------------------------------------------------
__global__ __launch_bounds__(256) void nrm_kernel(const float* __restrict__ src,
                                                  bf16_t* __restrict__ hi,
                                                  bf16_t* __restrict__ lo,
                                                  float* __restrict__ sq,
                                                  int rows) {
  int wave = (int)((blockIdx.x * blockDim.x + threadIdx.x) >> 5);
  int lane = (int)(threadIdx.x & 31);
  if (wave >= rows) return;
  const float* r = src + (size_t)wave * DIM;

  float vals[16];
  float ss = 0.0f;
#pragma unroll
  for (int i = 0; i < 16; ++i) {
    float v = r[lane + 32 * i];
    vals[i] = v;
    ss += v * v;
  }
#pragma unroll
  for (int off = 16; off >= 1; off >>= 1) ss += __shfl_xor(ss, off, 32);

  float inv = 1.0f / fmaxf(sqrtf(ss), 1e-12f);
  if (lane == 0) sq[wave] = ss * inv * inv;

  bf16_t* hrow = hi + (size_t)wave * DIM;
  bf16_t* lrow = lo + (size_t)wave * DIM;
#pragma unroll
  for (int i = 0; i < 16; ++i) {
    float x = vals[i] * inv;
    bf16_t h = (bf16_t)x;
    bf16_t l = (bf16_t)(x - (float)h);
    hrow[lane + 32 * i] = h;
    lrow[lane + 32 * i] = l;
  }
}

// ---------------------------------------------------------------------------
// Fragment load: documented 16-bit A-matrix 16x32 layout. Lane covers row
// (lane&15); elements 0..7 = K (kk + half*8 .. +7), 8..15 = K (+16 .. +23).
// Two contiguous 16B loads (global_load_b128 or ds_load_b128 after AS infer).
// ---------------------------------------------------------------------------
__device__ inline v16bf load_frag(const bf16_t* row, int k0) {
  v8bf c0 = *(const v8bf*)(row + k0);
  v8bf c1 = *(const v8bf*)(row + k0 + 16);
  return __builtin_shufflevector(c0, c1, 0, 1, 2, 3, 4, 5, 6, 7,
                                 8, 9, 10, 11, 12, 13, 14, 15);
}

// ---------------------------------------------------------------------------
// TDM 2D tile load: 80 rows x 512 bf16, row stride 512, global -> LDS.
// D# per CDNA5 ISA: group0 {count, lds_addr, gaddr,
// type=2}; group1 {data_size=2B, tensor_dim0=512, tensor_dim1=80,
// tile_dim0=512, tile_dim1=80, dim0_stride=512}; groups 2/3 zero (2D tile).
// This toolchain exposes the 6-arg builtin (extra zero-filled int32x8).
// ---------------------------------------------------------------------------
__device__ inline void tdm_load_panel(unsigned lds_addr,
                                      unsigned long long gaddr) {
  v4u g0 = { 1u,                                   // count=1, load, no gather
             lds_addr,                             // LDS byte address
             (unsigned)(gaddr & 0xffffffffu),      // global_addr[31:0]
             (unsigned)((gaddr >> 32) & 0x1ffffffu) | (2u << 30) }; // type=2
  v8i g1 = { (int)(1u << 16),        // workgroup_mask=0, data_size=1 (2B)
             (int)(512u << 16),      // tensor_dim0[15:0]=512 at bits[63:48]
             (int)(80u << 16),       // tensor_dim0 hi=0 | tensor_dim1=80
             (int)(512u << 16),      // tensor_dim1 hi=0 | tile_dim0=512
             (int)80,                // tile_dim1=80, tile_dim2=0
             (int)512,               // tensor_dim0_stride = 512 elements
             0, 0 };                 // stride hi, tensor_dim1_stride unused
  v4i z4 = { 0, 0, 0, 0 };
  v8i z8 = { 0, 0, 0, 0, 0, 0, 0, 0 };
  __builtin_amdgcn_tensor_load_to_lds(g0, g1, z4, z4, z8, 0);
}

// ---------------------------------------------------------------------------
// Phase 2: WMMA GEMM + fused exp/row-reduce.
// Block b -> (wgM = b/NGROUP, g = b%NGROUP): rows [128*wgM, +128) x cols
// [80*g, +80). Wave w owns M-tile wgM*8 + w; the 80x512 hi/lo B panel is
// TDM-staged into LDS once and shared by all 8 waves.
// ---------------------------------------------------------------------------
__global__ __launch_bounds__(256) void gemm_kernel(
    const bf16_t* __restrict__ fhi, const bf16_t* __restrict__ flo,
    const bf16_t* __restrict__ chi, const bf16_t* __restrict__ clo,
    const float* __restrict__ f2, const float* __restrict__ c2,
    float* __restrict__ part) {
  extern __shared__ char smem[];
  bf16_t* Bh = (bf16_t*)smem;
  bf16_t* Bl = (bf16_t*)(smem + BPANEL_BYTES);

  int wgM  = (int)blockIdx.x / NGROUP;     // 0..63  (M supertile of 128 rows)
  int g    = (int)blockIdx.x % NGROUP;     // 0..124 (column group of 80)
  int w    = (int)(threadIdx.x >> 5);      // wave in block = M-tile
  int lane = (int)(threadIdx.x & 31);
  int lane16 = lane & 15;
  int half   = lane >> 4;

  int mbase = wgM * (WAVES_PER_BLK * 16) + w * 16;
  int nbase = g * (16 * NTILE);

  // --- Stage B panel via Tensor Data Mover (wave 0 issues, all waves wait) --
  if (w == 0) {
    tdm_load_panel((unsigned)(uintptr_t)Bh,
                   (unsigned long long)(uintptr_t)(chi + (size_t)nbase * DIM));
    tdm_load_panel((unsigned)(uintptr_t)Bl,
                   (unsigned long long)(uintptr_t)(clo + (size_t)nbase * DIM));
    __builtin_amdgcn_s_wait_tensorcnt(0);
  }
  __syncthreads();

  const bf16_t* arow_h = fhi + (size_t)(mbase + lane16) * DIM;
  const bf16_t* arow_l = flo + (size_t)(mbase + lane16) * DIM;

  const bf16_t* brow_h[NTILE];
  const bf16_t* brow_l[NTILE];
#pragma unroll
  for (int t = 0; t < NTILE; ++t) {
    brow_h[t] = Bh + (size_t)(t * 16 + lane16) * DIM;
    brow_l[t] = Bl + (size_t)(t * 16 + lane16) * DIM;
  }

  v8f acc[NTILE];
#pragma unroll
  for (int t = 0; t < NTILE; ++t) acc[t] = (v8f){0, 0, 0, 0, 0, 0, 0, 0};

  for (int kk = 0; kk < DIM; kk += 32) {
    int k0 = kk + half * 8;
    v16bf ah = load_frag(arow_h, k0);
    v16bf al = load_frag(arow_l, k0);
#pragma unroll
    for (int t = 0; t < NTILE; ++t) {
      v16bf bh = load_frag(brow_h[t], k0);
      v16bf bl = load_frag(brow_l[t], k0);
      // bf16x3: hi*hi + hi*lo + lo*hi  (lo*lo negligible)
      acc[t] = __builtin_amdgcn_wmma_f32_16x16x32_bf16(
          false, ah, false, bh, (short)0, acc[t], false, false);
      acc[t] = __builtin_amdgcn_wmma_f32_16x16x32_bf16(
          false, ah, false, bl, (short)0, acc[t], false, false);
      acc[t] = __builtin_amdgcn_wmma_f32_16x16x32_bf16(
          false, al, false, bh, (short)0, acc[t], false, false);
    }
  }

  // Epilogue: dis = -5*(f2 + c2 - 2*dot); accumulate exp(dis) per row.
  float csq[NTILE];
#pragma unroll
  for (int t = 0; t < NTILE; ++t) csq[t] = c2[nbase + t * 16 + lane16];

#pragma unroll
  for (int r = 0; r < 8; ++r) {
    int m = mbase + half * 8 + r;   // C/D layout: VGPR r -> row r (+8 hi half)
    float fsq = f2[m];
    float esum = 0.0f;
#pragma unroll
    for (int t = 0; t < NTILE; ++t) {
      float dis = -SCALE * (fsq + csq[t] - 2.0f * acc[t][r]);
      esum += expf(dis);
    }
#pragma unroll
    for (int off = 8; off >= 1; off >>= 1) esum += __shfl_xor(esum, off, 32);
    if (lane16 == 0) part[(size_t)m * NGROUP + g] = esum;
  }
}

// ---------------------------------------------------------------------------
// Phase 3a: exact f32 positive-column dot per row (one wave per row).
// ---------------------------------------------------------------------------
__global__ __launch_bounds__(256) void pos_kernel(
    const bf16_t* __restrict__ fhi, const bf16_t* __restrict__ flo,
    const bf16_t* __restrict__ chi, const bf16_t* __restrict__ clo,
    const float* __restrict__ f2, const float* __restrict__ c2,
    const int* __restrict__ labels, const float* __restrict__ bias,
    float* __restrict__ dlab, float* __restrict__ posm) {
  int n = (int)((blockIdx.x * blockDim.x + threadIdx.x) >> 5);
  int lane = (int)(threadIdx.x & 31);
  if (n >= N_ROWS) return;
  int lab = labels[n];

  const bf16_t* fh = fhi + (size_t)n * DIM;
  const bf16_t* fl = flo + (size_t)n * DIM;
  const bf16_t* ch = chi + (size_t)lab * DIM;
  const bf16_t* cl = clo + (size_t)lab * DIM;

  float dot = 0.0f;
#pragma unroll
  for (int i = 0; i < 16; ++i) {
    int d = lane + 32 * i;
    float f = (float)fh[d] + (float)fl[d];
    float c = (float)ch[d] + (float)cl[d];
    dot = fmaf(f, c, dot);
  }
#pragma unroll
  for (int off = 16; off >= 1; off >>= 1) dot += __shfl_xor(dot, off, 32);

  if (lane == 0) {
    float d = -SCALE * (f2[n] + c2[lab] - 2.0f * dot);
    dlab[n] = d;
    posm[n] = d + bias[lab];
  }
}

// ---------------------------------------------------------------------------
// Phase 3b: fixed-order reduction of partial row sums (deterministic).
// ---------------------------------------------------------------------------
__global__ __launch_bounds__(256) void rowsum_kernel(const float* __restrict__ part,
                                                     float* __restrict__ sumexp) {
  int n = (int)(blockIdx.x * blockDim.x + threadIdx.x);
  if (n >= N_ROWS) return;
  const float* p = part + (size_t)n * NGROUP;
  float s = 0.0f;
  for (int gIdx = 0; gIdx < NGROUP; ++gIdx) s += p[gIdx];
  sumexp[n] = s;
}

// ---------------------------------------------------------------------------
// Phase 4: logits, mean, unbiased variance -> (loss, variance).
// ---------------------------------------------------------------------------
__global__ __launch_bounds__(1024) void final_kernel(
    const float* __restrict__ sumexp, const float* __restrict__ dlab,
    const float* __restrict__ posm, float* __restrict__ out) {
  __shared__ float sL[1024], sP[1024], sQ[1024];
  int tid = (int)threadIdx.x;
  float aL = 0.0f, aP = 0.0f, aQ = 0.0f;
  for (int n = tid; n < N_ROWS; n += 1024) {
    float pos = posm[n];
    float dl  = dlab[n];
    float denom = sumexp[n] - expf(dl) + expf(pos);
    aL += logf(denom) - pos;
    aP += pos;
    aQ += pos * pos;
  }
  sL[tid] = aL; sP[tid] = aP; sQ[tid] = aQ;
  __syncthreads();
  for (int off = 512; off >= 1; off >>= 1) {
    if (tid < off) {
      sL[tid] += sL[tid + off];
      sP[tid] += sP[tid + off];
      sQ[tid] += sQ[tid + off];
    }
    __syncthreads();
  }
  if (tid == 0) {
    const float Nf = (float)N_ROWS;
    float mean_logit = sL[0] / Nf;
    float var = (sQ[0] - sP[0] * sP[0] / Nf) / (Nf - 1.0f);
    out[0] = mean_logit + var;
    out[1] = var;
  }
}

// ---------------------------------------------------------------------------
extern "C" void kernel_launch(void* const* d_in, const int* in_sizes, int n_in,
                              void* d_out, int out_size, void* d_ws, size_t ws_size,
                              hipStream_t stream) {
  (void)in_sizes; (void)n_in; (void)out_size; (void)ws_size;
  const float* features = (const float*)d_in[0];
  const int*   labels   = (const int*)d_in[1];
  const float* centers  = (const float*)d_in[2];
  const float* bias     = (const float*)d_in[3];
  float* out = (float*)d_out;

  char* ws = (char*)d_ws;
  size_t off = 0;
  bf16_t* fhi = (bf16_t*)(ws + off); off += (size_t)N_ROWS * DIM * 2;
  bf16_t* flo = (bf16_t*)(ws + off); off += (size_t)N_ROWS * DIM * 2;
  bf16_t* chi = (bf16_t*)(ws + off); off += (size_t)N_CLS * DIM * 2;
  bf16_t* clo = (bf16_t*)(ws + off); off += (size_t)N_CLS * DIM * 2;
  float*  f2  = (float*)(ws + off);  off += (size_t)N_ROWS * 4;
  float*  c2  = (float*)(ws + off);  off += ((size_t)N_CLS * 4 + 255) & ~(size_t)255;
  float*  part= (float*)(ws + off);  off += (size_t)N_ROWS * NGROUP * 4;
  float*  sume= (float*)(ws + off);  off += (size_t)N_ROWS * 4;
  float*  dlab= (float*)(ws + off);  off += (size_t)N_ROWS * 4;
  float*  posm= (float*)(ws + off);  off += (size_t)N_ROWS * 4;
  // total ~41.5MB of workspace; every cell fully rewritten each call.

  // Allow 160KB dynamic LDS (CDNA5 WGP has 320KB). Idempotent host call.
  (void)hipFuncSetAttribute((const void*)gemm_kernel,
                            hipFuncAttributeMaxDynamicSharedMemorySize,
                            LDS_BYTES);

  // Phase 1: normalize (one wave per row)
  nrm_kernel<<<(N_ROWS + 7) / 8, 256, 0, stream>>>(features, fhi, flo, f2, N_ROWS);
  nrm_kernel<<<(N_CLS + 7) / 8, 256, 0, stream>>>(centers, chi, clo, c2, N_CLS);

  // Phase 2: TDM-staged WMMA GEMM + fused exp row-partials
  gemm_kernel<<<(MTILES / WAVES_PER_BLK) * NGROUP, 256, LDS_BYTES, stream>>>(
      fhi, flo, chi, clo, f2, c2, part);

  // Phase 3: positive column (exact f32) + deterministic row-sum reduce
  pos_kernel<<<(N_ROWS + 7) / 8, 256, 0, stream>>>(fhi, flo, chi, clo, f2, c2,
                                                   labels, bias, dlab, posm);
  rowsum_kernel<<<(N_ROWS + 255) / 256, 256, 0, stream>>>(part, sume);

  // Phase 4: scalar loss + variance
  final_kernel<<<1, 1024, 0, stream>>>(sume, dlab, posm, out);
}
